// SelfAttention1d_85152021611289
// MI455X (gfx1250) — matrix operand order repaired and checked by reading, not compiled, not run on hardware
//
#include <hip/hip_runtime.h>

// ---------------------------------------------------------------------------
// SelfAttention1d for MI455X (gfx1250): GroupNorm -> QKV -> flash-attn -> proj
// N=8, C=512, S=2048, H=8, D=64.  All GEMMs via v_wmma_f32_16x16x32_f16.
// ---------------------------------------------------------------------------

namespace {
constexpr int N_ = 8, C_ = 512, S_ = 2048, H_ = 8, D_ = 64;
constexpr float EPS_ = 1e-5f;
constexpr float QK_SCALE_ = 0.35355339059327373f;  // D^-0.25

typedef __attribute__((ext_vector_type(16))) _Float16 v16h;
typedef __attribute__((ext_vector_type(8)))  _Float16 v8h;
typedef __attribute__((ext_vector_type(4)))  _Float16 v4h;
typedef __attribute__((ext_vector_type(8)))  float    v8f;

// Build a 16-half fragment from two 8-half (16B-aligned) chunks.
__device__ __forceinline__ v16h ld_frag(const _Float16* p0, const _Float16* p1) {
  v8h a = *(const v8h*)p0;
  v8h b = *(const v8h*)p1;
  return __builtin_shufflevector(a, b, 0,1,2,3,4,5,6,7,8,9,10,11,12,13,14,15);
}

__device__ __forceinline__ v8f wmma16(v16h a, v16h b, v8f c) {
  // (neg_a, A, neg_b, B, c_mod, C, reuse_a, reuse_b)
  return __builtin_amdgcn_wmma_f32_16x16x32_f16(false, a, false, b, (short)0, c,
                                                false, false);
}

// Reductions across the 16-lane half-wave (masks 1,2,4,8 stay within a half).
__device__ __forceinline__ float redmax16(float v) {
  v = fmaxf(v, __shfl_xor(v, 8, 32));
  v = fmaxf(v, __shfl_xor(v, 4, 32));
  v = fmaxf(v, __shfl_xor(v, 2, 32));
  v = fmaxf(v, __shfl_xor(v, 1, 32));
  return v;
}
__device__ __forceinline__ float redsum16(float v) {
  v += __shfl_xor(v, 8, 32);
  v += __shfl_xor(v, 4, 32);
  v += __shfl_xor(v, 2, 32);
  v += __shfl_xor(v, 1, 32);
  return v;
}
}  // namespace

// ---------------------------------------------------------------------------
// K1: per-sample mean / rstd over C*S elements (GroupNorm with 1 group)
// ---------------------------------------------------------------------------
__global__ void __launch_bounds__(1024) gn_stats_kernel(
    const float* __restrict__ x, float* __restrict__ stats) {
  const int n = blockIdx.x;
  const float* xp = x + (size_t)n * C_ * S_;
  const int tid = threadIdx.x;
  float s = 0.f, s2 = 0.f;
  const int total4 = C_ * S_ / 4;
  for (int i = tid; i < total4; i += 1024) {
    float4 v = ((const float4*)xp)[i];
    s  += v.x + v.y + v.z + v.w;
    s2 += v.x * v.x + v.y * v.y + v.z * v.z + v.w * v.w;
  }
  for (int m = 16; m >= 1; m >>= 1) {
    s += __shfl_xor(s, m, 32);
    s2 += __shfl_xor(s2, m, 32);
  }
  __shared__ float ls[32], ls2[32];
  const int wv = tid >> 5, ln = tid & 31;
  if (ln == 0) { ls[wv] = s; ls2[wv] = s2; }
  __syncthreads();
  if (wv == 0) {
    s = ls[ln]; s2 = ls2[ln];
    for (int m = 16; m >= 1; m >>= 1) {
      s += __shfl_xor(s, m, 32);
      s2 += __shfl_xor(s2, m, 32);
    }
    if (ln == 0) {
      const float inv = 1.0f / (float)(C_ * S_);
      const float mean = s * inv;
      const float var = s2 * inv - mean * mean;
      stats[2 * n]     = mean;
      stats[2 * n + 1] = rsqrtf(var + EPS_);
    }
  }
}

// ---------------------------------------------------------------------------
// K0: f32 -> f16 weight conversion (4 elems / thread)
// ---------------------------------------------------------------------------
__global__ void cvt_f16_kernel(const float* __restrict__ src,
                               _Float16* __restrict__ dst, int n4) {
  const int i = blockIdx.x * blockDim.x + threadIdx.x;
  if (i < n4) {
    float4 v = ((const float4*)src)[i];
    v4h o = {(_Float16)v.x, (_Float16)v.y, (_Float16)v.z, (_Float16)v.w};
    ((v4h*)dst)[i] = o;
  }
}

// ---------------------------------------------------------------------------
// K2: QKV GEMM.  Block = 128 thr (4 waves), tile = 64(o) x 16(s), K loop = 512.
// B tile (32c x 16s) is normalized + transposed into LDS [s][c] so the WMMA
// B-fragment is a contiguous per-lane 32B read.  Outputs:
//   q,k : f16 [n,h,S,D] with D^-0.25 folded in
//   v   : f16 [n,h,D,S] (transposed for the p*v B-fragment later)
// ---------------------------------------------------------------------------
__global__ void __launch_bounds__(128) qkv_gemm_kernel(
    const float* __restrict__ x, const float* __restrict__ stats,
    const float* __restrict__ gn_w, const float* __restrict__ gn_b,
    const _Float16* __restrict__ w16, const float* __restrict__ qkv_b,
    _Float16* __restrict__ qO, _Float16* __restrict__ kO,
    _Float16* __restrict__ vT) {
  constexpr int BS = 40;  // padded row stride (halfwords): bank-spread, 16B-aligned
  __shared__ _Float16 ldsB[16 * BS];
  const int blk   = blockIdx.x;
  const int n     = blk / (24 * 128);
  const int rem   = blk % (24 * 128);
  const int obase = (rem / 128) * 64;
  const int s0    = (rem % 128) * 16;
  const int tid   = threadIdx.x;
  const int wv    = tid >> 5;
  const int lane  = tid & 31;
  const int half  = lane >> 4;
  const int l15   = lane & 15;
  const int ow    = obase + 16 * wv;
  const int k0a   = half ? 16 : 0;  // A-fragment K chunk base
  const int klo   = half ? 16 : 0;  // B-fragment K chunk base

  const float mean = stats[2 * n];
  const float rstd = stats[2 * n + 1];
  const float* xp = x + (size_t)n * C_ * S_;

  const int crow_l = tid >> 2;       // 0..31 : c within tile
  const int soff   = (tid & 3) * 4;  // 0,4,8,12 : s within tile

  v8f acc = {};
  for (int cbase = 0; cbase < C_; cbase += 32) {
    const int c = cbase + crow_l;
    const float gw = gn_w[c] * rstd;
    const float gb = gn_b[c] - mean * gw;  // xn = x*gw + gb
    float4 v = *(const float4*)(xp + (size_t)c * S_ + s0 + soff);
    __syncthreads();  // previous iter's fragment reads done
    ldsB[(soff + 0) * BS + crow_l] = (_Float16)(v.x * gw + gb);
    ldsB[(soff + 1) * BS + crow_l] = (_Float16)(v.y * gw + gb);
    ldsB[(soff + 2) * BS + crow_l] = (_Float16)(v.z * gw + gb);
    ldsB[(soff + 3) * BS + crow_l] = (_Float16)(v.w * gw + gb);
    __syncthreads();
    const _Float16* ap = w16 + (size_t)(ow + l15) * C_ + cbase + k0a;
    const _Float16* bp = &ldsB[l15 * BS + klo];
    acc = wmma16(ld_frag(ap, ap + 8), ld_frag(bp, bp + 8), acc);
  }

  // Epilogue: bias, scale, layout-aware store.
  const int region = ow >> 9;               // 0=q 1=k 2=v
  const int o_l    = ow & 511;
  const int h      = o_l >> 6;
  const int dbase  = (o_l & 63) + 8 * half; // lane's 8 consecutive d values
  if (region < 2) {
    v8h outv;
#pragma unroll
    for (int r = 0; r < 8; ++r)
      outv[r] = (_Float16)(QK_SCALE_ * (acc[r] + qkv_b[ow + 8 * half + r]));
    _Float16* dst = (region == 0 ? qO : kO) +
                    ((size_t)(n * H_ + h) * S_ + s0 + l15) * D_ + dbase;
    *(v8h*)dst = outv;  // 16B packed store
  } else {
#pragma unroll
    for (int r = 0; r < 8; ++r) {
      const float val = acc[r] + qkv_b[ow + 8 * half + r];
      vT[((size_t)(n * H_ + h) * D_ + dbase + r) * S_ + s0 + l15] =
          (_Float16)val;
    }
  }
}

// ---------------------------------------------------------------------------
// K3: fused flash attention.  One wave per 16-row q-tile; loop over 32 keys
// per step: 4 score WMMAs, online softmax (half-wave shfl reductions matching
// the C-layout row mapping), p -> A-layout via padded LDS, 4 p*v WMMAs.
// ---------------------------------------------------------------------------
__global__ void __launch_bounds__(32) attn_kernel(
    const _Float16* __restrict__ q, const _Float16* __restrict__ k,
    const _Float16* __restrict__ vT, _Float16* __restrict__ y) {
  constexpr int PS = 40, YS = 72;  // padded LDS row strides (halfwords)
  __shared__ _Float16 ldsP[16 * PS];
  __shared__ _Float16 ldsY[16 * YS];
  const int blk  = blockIdx.x;
  const int nh   = blk / (S_ / 16);
  const int s0   = (blk % (S_ / 16)) * 16;
  const int lane = threadIdx.x;
  const int half = lane >> 4;
  const int l15  = lane & 15;
  const int k0a  = half ? 16 : 0;
  const int klo  = half ? 16 : 0;

  const _Float16* qb = q + (size_t)nh * S_ * D_;
  const _Float16* kb = k + (size_t)nh * S_ * D_;
  const _Float16* vb = vT + (size_t)nh * D_ * S_;

  // Resident q A-fragments: d 0..31 and 32..63.
  const _Float16* qp = qb + (size_t)(s0 + l15) * D_;
  const v16h qa0 = ld_frag(qp + k0a, qp + k0a + 8);
  const v16h qa1 = ld_frag(qp + 32 + k0a, qp + 32 + k0a + 8);

  v8f y0 = {}, y1 = {}, y2 = {}, y3 = {};  // 16 q-rows x 64 d accumulators
  float mrow[8], lrow[8];
#pragma unroll
  for (int r = 0; r < 8; ++r) { mrow[r] = -3.0e38f; lrow[r] = 0.f; }

  for (int kc = 0; kc < S_; kc += 32) {
    // --- scores: two 16-key tiles, contraction over d (2 x K=32) ---
    v8f sc0 = {}, sc1 = {};
    {
      const _Float16* kp = kb + (size_t)(kc + l15) * D_;
      sc0 = wmma16(qa0, ld_frag(kp + klo, kp + klo + 8), sc0);
      sc0 = wmma16(qa1, ld_frag(kp + 32 + klo, kp + 32 + klo + 8), sc0);
    }
    {
      const _Float16* kp = kb + (size_t)(kc + 16 + l15) * D_;
      sc1 = wmma16(qa0, ld_frag(kp + klo, kp + klo + 8), sc1);
      sc1 = wmma16(qa1, ld_frag(kp + 32 + klo, kp + 32 + klo + 8), sc1);
    }
    // --- online softmax per row (C-layout: VGPR r holds row r + 8*half) ---
#pragma unroll
    for (int r = 0; r < 8; ++r) {
      const float mx = redmax16(fmaxf(sc0[r], sc1[r]));
      const float mn = fmaxf(mrow[r], mx);
      const float al = __expf(mrow[r] - mn);
      const float p0 = __expf(sc0[r] - mn);
      const float p1 = __expf(sc1[r] - mn);
      lrow[r] = lrow[r] * al + redsum16(p0 + p1);
      mrow[r] = mn;
      y0[r] *= al; y1[r] *= al; y2[r] *= al; y3[r] *= al;
      ldsP[(r + 8 * half) * PS + l15]      = (_Float16)p0;
      ldsP[(r + 8 * half) * PS + 16 + l15] = (_Float16)p1;
    }
    __syncthreads();  // single-wave WG: compiler ordering + dscnt wait
    const _Float16* pp = &ldsP[l15 * PS + k0a];
    const v16h pa = ld_frag(pp, pp + 8);  // p in A-layout, K=32 keys
    // --- y += p * v : 4 d-tiles, v B-fragments contiguous from vT[d][s] ---
    {
      const _Float16* vp = vb + (size_t)(l15) * S_ + kc + klo;
      y0 = wmma16(pa, ld_frag(vp, vp + 8), y0);
    }
    {
      const _Float16* vp = vb + (size_t)(16 + l15) * S_ + kc + klo;
      y1 = wmma16(pa, ld_frag(vp, vp + 8), y1);
    }
    {
      const _Float16* vp = vb + (size_t)(32 + l15) * S_ + kc + klo;
      y2 = wmma16(pa, ld_frag(vp, vp + 8), y2);
    }
    {
      const _Float16* vp = vb + (size_t)(48 + l15) * S_ + kc + klo;
      y3 = wmma16(pa, ld_frag(vp, vp + 8), y3);
    }
    __syncthreads();
  }

  // Epilogue: normalize, transpose through LDS, coalesced f16 writes.
#pragma unroll
  for (int r = 0; r < 8; ++r) {
    const float inv = 1.0f / lrow[r];
    const int row = (r + 8 * half) * YS;
    ldsY[row + l15]      = (_Float16)(y0[r] * inv);
    ldsY[row + 16 + l15] = (_Float16)(y1[r] * inv);
    ldsY[row + 32 + l15] = (_Float16)(y2[r] * inv);
    ldsY[row + 48 + l15] = (_Float16)(y3[r] * inv);
  }
  __syncthreads();
  const _Float16* src = &ldsY[l15 * YS + half * 32];
  _Float16* dst = y + ((size_t)nh * S_ + s0 + l15) * D_ + half * 32;
#pragma unroll
  for (int j = 0; j < 4; ++j)
    *(v8h*)(dst + 8 * j) = *(const v8h*)(src + 8 * j);
}

// ---------------------------------------------------------------------------
// K4: out projection + bias + residual.  One wave per 16(o) x 16(s) tile.
// B fragments load directly from y[n,h,S,D] (per-lane contiguous K chunk).
// ---------------------------------------------------------------------------
__global__ void __launch_bounds__(256) outproj_kernel(
    const float* __restrict__ inpt, const _Float16* __restrict__ w16,
    const float* __restrict__ out_b, const _Float16* __restrict__ y,
    float* __restrict__ out) {
  const int gw   = blockIdx.x * 8 + ((int)threadIdx.x >> 5);
  const int lane = threadIdx.x & 31;
  const int half = lane >> 4;
  const int l15  = lane & 15;
  const int k0a  = half ? 16 : 0;
  const int klo  = half ? 16 : 0;
  const int n     = gw / (32 * 128);
  const int rem   = gw % (32 * 128);
  const int obase = (rem / 128) * 16;
  const int s0    = (rem % 128) * 16;

  v8f acc = {};
#pragma unroll 4
  for (int kk = 0; kk < 16; ++kk) {
    const int cbase = kk * 32;
    const _Float16* ap = w16 + (size_t)(obase + l15) * C_ + cbase + k0a;
    const int h  = cbase >> 6;
    const int d0 = (cbase & 63) + klo;
    const _Float16* bp =
        y + ((size_t)(n * H_ + h) * S_ + s0 + l15) * D_ + d0;
    acc = wmma16(ld_frag(ap, ap + 8), ld_frag(bp, bp + 8), acc);
  }
#pragma unroll
  for (int r = 0; r < 8; ++r) {
    const int o = obase + r + 8 * half;
    const size_t idx = ((size_t)n * C_ + o) * S_ + s0 + l15;
    out[idx] = inpt[idx] + out_b[o] + acc[r];
  }
}

// ---------------------------------------------------------------------------
extern "C" void kernel_launch(void* const* d_in, const int* in_sizes, int n_in,
                              void* d_out, int out_size, void* d_ws,
                              size_t ws_size, hipStream_t stream) {
  (void)in_sizes; (void)n_in; (void)out_size; (void)ws_size;
  const float* inpt  = (const float*)d_in[0];
  const float* gn_w  = (const float*)d_in[1];
  const float* gn_b  = (const float*)d_in[2];
  const float* qkv_w = (const float*)d_in[3];
  const float* qkv_b = (const float*)d_in[4];
  const float* out_w = (const float*)d_in[5];
  const float* out_b = (const float*)d_in[6];
  float* out = (float*)d_out;

  // Workspace layout (~69.2 MB):
  //   [0,256)            : per-sample mean/rstd
  //   wq16 : 3*C*C f16   (1.5 MB)
  //   wo16 : C*C   f16   (0.5 MB)
  //   q,k  : [N,H,S,D] f16 (16 MB each)
  //   vT   : [N,H,D,S] f16 (16 MB)
  //   y    : [N,H,S,D] f16 (16 MB)
  char* ws = (char*)d_ws;
  float*    stats = (float*)ws;
  _Float16* wq16  = (_Float16*)(ws + 256);
  _Float16* wo16  = wq16 + (size_t)3 * C_ * C_;
  _Float16* qB    = wo16 + (size_t)C_ * C_;
  const size_t HS = (size_t)N_ * H_ * S_ * D_;
  _Float16* kB = qB + HS;
  _Float16* vT = kB + HS;
  _Float16* yB = vT + HS;

  cvt_f16_kernel<<<(3 * C_ * C_ / 4 + 255) / 256, 256, 0, stream>>>(
      qkv_w, wq16, 3 * C_ * C_ / 4);
  cvt_f16_kernel<<<(C_ * C_ / 4 + 255) / 256, 256, 0, stream>>>(
      out_w, wo16, C_ * C_ / 4);
  gn_stats_kernel<<<N_, 1024, 0, stream>>>(inpt, stats);
  qkv_gemm_kernel<<<N_ * 24 * 128, 128, 0, stream>>>(
      inpt, stats, gn_w, gn_b, wq16, qkv_b, qB, kB, vT);
  attn_kernel<<<N_ * H_ * (S_ / 16), 32, 0, stream>>>(qB, kB, vT, yB);
  outproj_kernel<<<N_ * 32 * 128 / 8, 256, 0, stream>>>(
      inpt, wo16, out_b, yB, out);
}
